// InverterNet_82643760709740
// MI455X (gfx1250) — matrix operand order repaired and checked by reading, not compiled, not run on hardware
//
#include <hip/hip_runtime.h>
#include <hip/hip_bf16.h>

typedef __attribute__((ext_vector_type(16))) _Float16 v16h;
typedef __attribute__((ext_vector_type(8)))  _Float16 h8;
typedef __attribute__((ext_vector_type(8)))  float    v8f;

#define CDIM 3
#define FDIM 32
#define HDIM 64
#define MDIM 32
#define XDIM (CDIM + FDIM)   // 35 floats per node row

// silu via hardware reciprocal (v_rcp_f32), avoids IEEE division expansion
__device__ __forceinline__ float silu_f(float v) {
  return v * __builtin_amdgcn_rcpf(1.0f + __expf(-v));
}

// ---- WMMA fragment helpers (CDNA5 ISA 7.12.2 layouts, wave32) -------------
// A (16xK f16 tile, row-major in LDS, stride lda halves), K-offset kk.
// lanes 0-15: halves 0..7 -> K kk+0..7,   halves 8..15 -> K kk+16..23
// lanes16-31: halves 0..7 -> K kk+8..15,  halves 8..15 -> K kk+24..31
__device__ __forceinline__ v16h load_a_frag(const _Float16* sm, int lda, int kk) {
  const int lane = threadIdx.x & 31;
  const int m  = lane & 15;
  const int hi = lane >> 4;
  const h8 a0 = *((const h8*)(sm + m * lda + kk + hi * 8));
  const h8 a1 = *((const h8*)(sm + m * lda + kk + 16 + hi * 8));
  return __builtin_shufflevector(a0, a1, 0, 1, 2, 3, 4, 5, 6, 7,
                                 8, 9, 10, 11, 12, 13, 14, 15);
}

// B fragments pre-swizzled in LDS: (tile, lane) -> 16 contiguous halves.
__device__ __forceinline__ v16h load_b_swz(const _Float16* base, int tile) {
  const int lane = threadIdx.x & 31;
  return *((const v16h*)(base + (tile * 32 + lane) * 16));
}

// Stage KxN row-major f32 weights into swizzled-f16 B-fragment order.
template <int K, int N, int KPAD>
__device__ __forceinline__ void stage_b_swz(_Float16* dst,
                                            const float* __restrict__ W) {
  constexpr int tilesN = N / 16;
  for (int idx = threadIdx.x; idx < KPAD * N; idx += 256) {
    const int k  = idx / N;
    const int nn = idx - k * N;
    const float v = (k < K) ? W[idx] : 0.0f;
    const int tile = (k >> 5) * tilesN + (nn >> 4);
    const int ln   = ((k >> 3) & 1) * 16 + (nn & 15);
    dst[(tile * 32 + ln) * 16 + ((k & 7) + ((k >> 1) & 8))] = (_Float16)v;
  }
}
// C/D fragment: element g of v8f <-> (m = g + 8*hi, n = lane&15)

// ---------------------------------------------------------------------------
__global__ void zero_kernel(float* __restrict__ p, int n) {
  int i = blockIdx.x * blockDim.x + threadIdx.x;
  if (i < n) p[i] = 0.0f;
}

// ---------------------------------------------------------------------------
// Edge kernel: 8 waves/block, 16 edges per wave (128 edges per block).
__global__ __launch_bounds__(256, 2) void egnn_edge_kernel(
    const float* __restrict__ x, const long long* __restrict__ ei,
    const float* __restrict__ ea,
    const float* __restrict__ We1, const float* __restrict__ be1,
    const float* __restrict__ We2, const float* __restrict__ be2,
    const float* __restrict__ Wc1, const float* __restrict__ bc1,
    const float* __restrict__ Wc2, const float* __restrict__ bc2,
    float* __restrict__ cnt, float* __restrict__ mhat, float* __restrict__ msum,
    int nE)
{
  __shared__ __attribute__((aligned(32))) _Float16 sWe1s[12 * 32 * 16]; // 96x64
  __shared__ __attribute__((aligned(32))) _Float16 sWe2s[4 * 32 * 16];  // 64x32
  __shared__ __attribute__((aligned(32))) _Float16 sWc1s[4 * 32 * 16];  // 32x64
  __shared__ float sWc2[64];
  __shared__ float sbe1[64], sbe2[32], sbc1[64];
  __shared__ __attribute__((aligned(32))) _Float16 sA[8 * 16 * 96];
  __shared__ int   sDst[8 * 16];
  __shared__ float sDiff[8 * 16 * 3];
  __shared__ float sCw[8 * 16];

  const int wave = threadIdx.x >> 5;
  const int lane = threadIdx.x & 31;
  const int n  = lane & 15;
  const int hi = lane >> 4;
  _Float16* sAw = sA + wave * 16 * 96;

  // ---- stage weights (f32 -> f16, B-fragment swizzle) ---------------------
  stage_b_swz<66, 64, 96>(sWe1s, We1);
  stage_b_swz<64, 32, 64>(sWe2s, We2);
  stage_b_swz<32, 64, 32>(sWc1s, Wc1);
  if (threadIdx.x < 64) {
    sWc2[threadIdx.x] = Wc2[threadIdx.x];
    sbe1[threadIdx.x] = be1[threadIdx.x];
    sbc1[threadIdx.x] = bc1[threadIdx.x];
  }
  if (threadIdx.x < 32) sbe2[threadIdx.x] = be2[threadIdx.x];

  // ---- phase 1: gather edge inputs into A tile ----------------------------
  {
    const int r    = lane >> 1;   // edge row within wave tile (2 lanes/edge)
    const int half = lane & 1;
    const int e  = blockIdx.x * 128 + wave * 16 + r;
    const bool ev = e < nE;
    const int ec = ev ? e : (nE - 1);
    const int s = (int)ei[ec];
    const int d = (int)ei[nE + ec];
    const float* xs = x + s * XDIM;
    const float* xd = x + d * XDIM;
    h8 vi0, vi1, vj0, vj1;
#pragma unroll
    for (int j = 0; j < 8; ++j) {
      vi0[j] = (_Float16)xd[CDIM + half * 16 + j];       // x_i = feats[dst]
      vi1[j] = (_Float16)xd[CDIM + half * 16 + 8 + j];
      vj0[j] = (_Float16)xs[CDIM + half * 16 + j];       // x_j = feats[src]
      vj1[j] = (_Float16)xs[CDIM + half * 16 + 8 + j];
    }
    *((h8*)(sAw + r * 96 + half * 16))      = vi0;
    *((h8*)(sAw + r * 96 + half * 16 + 8))  = vi1;
    *((h8*)(sAw + r * 96 + 32 + half * 16))     = vj0;
    *((h8*)(sAw + r * 96 + 32 + half * 16 + 8)) = vj1;
    if (half == 0) {
      float n2 = 0.0f;
#pragma unroll
      for (int c = 0; c < 3; ++c) {
        float dc = xs[c] - xd[c];
        sDiff[(wave * 16 + r) * 3 + c] = dc;
        n2 += dc * dc;
      }
      sAw[r * 96 + 64] = (_Float16)ea[ec];
      sAw[r * 96 + 65] = (_Float16)n2;
#pragma unroll
      for (int j = 66; j < 72; ++j) sAw[r * 96 + j] = (_Float16)0.0f;
      sDst[wave * 16 + r] = ev ? d : -1;
    } else {
      const h8 z = (h8)(_Float16)0.0f;
      *((h8*)(sAw + r * 96 + 72)) = z;
      *((h8*)(sAw + r * 96 + 80)) = z;
      *((h8*)(sAw + r * 96 + 88)) = z;
    }
  }
  __syncthreads();

  // ---- phase 2: h = silu(A(16x66) @ We1 + be1)  [12 WMMAs] ----------------
  v8f acc1[4];
  {
    v16h a = load_a_frag(sAw, 96, 0);
#pragma unroll
    for (int nt = 0; nt < 4; ++nt) {
      v16h b = load_b_swz(sWe1s, nt);
      acc1[nt] = __builtin_amdgcn_wmma_f32_16x16x32_f16(
          false, a, false, b, (short)0, (v8f){}, false, false);  // inline C=0
    }
  }
#pragma unroll
  for (int kt = 1; kt < 3; ++kt) {
    v16h a = load_a_frag(sAw, 96, kt * 32);
#pragma unroll
    for (int nt = 0; nt < 4; ++nt) {
      v16h b = load_b_swz(sWe1s, kt * 4 + nt);
      acc1[nt] = __builtin_amdgcn_wmma_f32_16x16x32_f16(
          false, a, false, b, (short)0, acc1[nt], false, false);
    }
  }
  __syncthreads();
#pragma unroll
  for (int nt = 0; nt < 4; ++nt) {
    const float b = sbe1[nt * 16 + n];
#pragma unroll
    for (int g = 0; g < 8; ++g)
      sAw[(g + 8 * hi) * 96 + nt * 16 + n] = (_Float16)silu_f(acc1[nt][g] + b);
  }
  __syncthreads();

  // ---- phase 3: m_ij = silu(h @ We2 + be2)  [4 WMMAs] ---------------------
  v8f acc2[2];
  {
    v16h a = load_a_frag(sAw, 96, 0);
#pragma unroll
    for (int nt = 0; nt < 2; ++nt) {
      v16h b = load_b_swz(sWe2s, nt);
      acc2[nt] = __builtin_amdgcn_wmma_f32_16x16x32_f16(
          false, a, false, b, (short)0, (v8f){}, false, false);
    }
  }
  {
    v16h a = load_a_frag(sAw, 96, 32);
#pragma unroll
    for (int nt = 0; nt < 2; ++nt) {
      v16h b = load_b_swz(sWe2s, 2 + nt);
      acc2[nt] = __builtin_amdgcn_wmma_f32_16x16x32_f16(
          false, a, false, b, (short)0, acc2[nt], false, false);
    }
  }
  float mreg[2][8];
#pragma unroll
  for (int nt = 0; nt < 2; ++nt) {
    const float b = sbe2[nt * 16 + n];
#pragma unroll
    for (int g = 0; g < 8; ++g) mreg[nt][g] = silu_f(acc2[nt][g] + b);
  }
  __syncthreads();
#pragma unroll
  for (int nt = 0; nt < 2; ++nt)
#pragma unroll
    for (int g = 0; g < 8; ++g)
      sAw[(g + 8 * hi) * 96 + nt * 16 + n] = (_Float16)mreg[nt][g];
  __syncthreads();

  // ---- phase 4: cw = silu(m_ij @ Wc1 + bc1) @ Wc2 + bc2  [4 WMMAs] --------
  v8f acc3[4];
  {
    v16h a = load_a_frag(sAw, 96, 0);
#pragma unroll
    for (int nt = 0; nt < 4; ++nt) {
      v16h b = load_b_swz(sWc1s, nt);
      acc3[nt] = __builtin_amdgcn_wmma_f32_16x16x32_f16(
          false, a, false, b, (short)0, (v8f){}, false, false);
    }
  }
  float p[8];
#pragma unroll
  for (int g = 0; g < 8; ++g) p[g] = 0.0f;
#pragma unroll
  for (int nt = 0; nt < 4; ++nt) {
    const float bb = sbc1[nt * 16 + n];
    const float w2 = sWc2[nt * 16 + n];
#pragma unroll
    for (int g = 0; g < 8; ++g) p[g] += silu_f(acc3[nt][g] + bb) * w2;
  }
#pragma unroll
  for (int off = 1; off < 16; off <<= 1)
#pragma unroll
    for (int g = 0; g < 8; ++g) p[g] += __shfl_xor(p[g], off, 32);
  const float bc2v = bc2[0];
  if (n == 0) {
#pragma unroll
    for (int g = 0; g < 8; ++g) sCw[wave * 16 + hi * 8 + g] = p[g] + bc2v;
  }
  __syncthreads();

  // ---- phase 5: scatter-add (atomics from C-fragment layout) --------------
#pragma unroll
  for (int nt = 0; nt < 2; ++nt)
#pragma unroll
    for (int g = 0; g < 8; ++g) {
      const int m = g + 8 * hi;
      const int d = sDst[wave * 16 + m];
      if (d >= 0) atomicAdd(&msum[d * MDIM + nt * 16 + n], mreg[nt][g]);
    }
  if (lane < 16) {
    const int m = lane;
    const int d = sDst[wave * 16 + m];
    if (d >= 0) {
      const float c = sCw[wave * 16 + m];
      atomicAdd(&cnt[d], 1.0f);
#pragma unroll
      for (int c3 = 0; c3 < 3; ++c3)
        atomicAdd(&mhat[d * 3 + c3], sDiff[(wave * 16 + m) * 3 + c3] * c);
    }
  }
}

// ---------------------------------------------------------------------------
// Node kernel: 8 waves/block, 16 nodes per wave (128 nodes per block).
__global__ __launch_bounds__(256, 2) void egnn_node_kernel(
    const float* __restrict__ x,
    const float* __restrict__ Wl1, const float* __restrict__ bl1,
    const float* __restrict__ Wl2, const float* __restrict__ bl2,
    const float* __restrict__ cnt, const float* __restrict__ mhat,
    const float* __restrict__ msum,
    float* __restrict__ out, int nN)
{
  __shared__ __attribute__((aligned(32))) _Float16 sWl1s[8 * 32 * 16]; // 64x64
  __shared__ __attribute__((aligned(32))) _Float16 sWl2s[4 * 32 * 16]; // 64x32
  __shared__ float sbl1[64], sbl2[32];
  __shared__ __attribute__((aligned(32))) _Float16 sA[8 * 16 * 64];

  const int wave = threadIdx.x >> 5;
  const int lane = threadIdx.x & 31;
  const int n  = lane & 15;
  const int hi = lane >> 4;
  const int base = blockIdx.x * 128 + wave * 16;
  _Float16* sAw = sA + wave * 16 * 64;

  stage_b_swz<64, 64, 64>(sWl1s, Wl1);
  stage_b_swz<64, 32, 64>(sWl2s, Wl2);
  if (threadIdx.x < 64) sbl1[threadIdx.x] = bl1[threadIdx.x];
  if (threadIdx.x < 32) sbl2[threadIdx.x] = bl2[threadIdx.x];

  // ---- phase 1: A = [feats | m_i], write coord_out ------------------------
  {
    const int r    = lane >> 1;
    const int half = lane & 1;
    const int i  = base + r;
    const bool v = i < nN;
    const int ic = v ? i : (nN - 1);
    const float dn = fmaxf(cnt[ic], 1.0f);
    const float rdn = __builtin_amdgcn_rcpf(dn);
    h8 f0, f1, m0, m1;
#pragma unroll
    for (int j = 0; j < 8; ++j) {
      f0[j] = (_Float16)x[ic * XDIM + CDIM + half * 16 + j];
      f1[j] = (_Float16)x[ic * XDIM + CDIM + half * 16 + 8 + j];
      m0[j] = (_Float16)(msum[ic * MDIM + half * 16 + j] * rdn);
      m1[j] = (_Float16)(msum[ic * MDIM + half * 16 + 8 + j] * rdn);
    }
    *((h8*)(sAw + r * 64 + half * 16))          = f0;
    *((h8*)(sAw + r * 64 + half * 16 + 8))      = f1;
    *((h8*)(sAw + r * 64 + 32 + half * 16))     = m0;
    *((h8*)(sAw + r * 64 + 32 + half * 16 + 8)) = m1;
    if (half == 0 && v) {
#pragma unroll
      for (int c = 0; c < 3; ++c)
        out[i * XDIM + c] = x[i * XDIM + c] + mhat[i * 3 + c] * rdn;
    }
  }
  __syncthreads();

  // ---- phase 2: h = silu(A @ Wl1 + bl1)  [8 WMMAs] ------------------------
  v8f acc[4];
  {
    v16h a = load_a_frag(sAw, 64, 0);
#pragma unroll
    for (int nt = 0; nt < 4; ++nt) {
      v16h b = load_b_swz(sWl1s, nt);
      acc[nt] = __builtin_amdgcn_wmma_f32_16x16x32_f16(
          false, a, false, b, (short)0, (v8f){}, false, false);
    }
  }
  {
    v16h a = load_a_frag(sAw, 64, 32);
#pragma unroll
    for (int nt = 0; nt < 4; ++nt) {
      v16h b = load_b_swz(sWl1s, 4 + nt);
      acc[nt] = __builtin_amdgcn_wmma_f32_16x16x32_f16(
          false, a, false, b, (short)0, acc[nt], false, false);
    }
  }
  __syncthreads();
#pragma unroll
  for (int nt = 0; nt < 4; ++nt) {
    const float b = sbl1[nt * 16 + n];
#pragma unroll
    for (int g = 0; g < 8; ++g)
      sAw[(g + 8 * hi) * 64 + nt * 16 + n] = (_Float16)silu_f(acc[nt][g] + b);
  }
  __syncthreads();

  // ---- phase 3: lat = h @ Wl2 + bl2, residual store  [4 WMMAs] ------------
  v8f acc2[2];
  {
    v16h a = load_a_frag(sAw, 64, 0);
#pragma unroll
    for (int nt = 0; nt < 2; ++nt) {
      v16h b = load_b_swz(sWl2s, nt);
      acc2[nt] = __builtin_amdgcn_wmma_f32_16x16x32_f16(
          false, a, false, b, (short)0, (v8f){}, false, false);
    }
  }
  {
    v16h a = load_a_frag(sAw, 64, 32);
#pragma unroll
    for (int nt = 0; nt < 2; ++nt) {
      v16h b = load_b_swz(sWl2s, 2 + nt);
      acc2[nt] = __builtin_amdgcn_wmma_f32_16x16x32_f16(
          false, a, false, b, (short)0, acc2[nt], false, false);
    }
  }
#pragma unroll
  for (int nt = 0; nt < 2; ++nt) {
    const float b = sbl2[nt * 16 + n];
#pragma unroll
    for (int g = 0; g < 8; ++g) {
      const int m = g + 8 * hi;
      const int i = base + m;
      if (i < nN) {
        const int j = nt * 16 + n;
        out[i * XDIM + CDIM + j] = x[i * XDIM + CDIM + j] + acc2[nt][g] + b;
      }
    }
  }
}

// ---------------------------------------------------------------------------
extern "C" void kernel_launch(void* const* d_in, const int* in_sizes, int n_in,
                              void* d_out, int out_size, void* d_ws, size_t ws_size,
                              hipStream_t stream) {
  const float*      x   = (const float*)d_in[0];
  const long long*  ei  = (const long long*)d_in[1];   // int64 edge_index
  const float*      ea  = (const float*)d_in[2];
  const float* We1 = (const float*)d_in[3];
  const float* be1 = (const float*)d_in[4];
  const float* We2 = (const float*)d_in[5];
  const float* be2 = (const float*)d_in[6];
  const float* Wc1 = (const float*)d_in[7];
  const float* bc1 = (const float*)d_in[8];
  const float* Wc2 = (const float*)d_in[9];
  const float* bc2 = (const float*)d_in[10];
  const float* Wl1 = (const float*)d_in[11];
  const float* bl1 = (const float*)d_in[12];
  const float* Wl2 = (const float*)d_in[13];
  const float* bl2 = (const float*)d_in[14];
  float* out = (float*)d_out;

  const int nN = in_sizes[0] / XDIM;
  const int nE = in_sizes[2];

  float* ws   = (float*)d_ws;
  float* cnt  = ws;               // N
  float* mhat = ws + nN;          // 3N
  float* msum = ws + 4 * nN;      // 32N

  const int zn = 36 * nN;
  zero_kernel<<<(zn + 255) / 256, 256, 0, stream>>>(ws, zn);

  egnn_edge_kernel<<<(nE + 127) / 128, 256, 0, stream>>>(
      x, ei, ea, We1, be1, We2, be2, Wc1, bc1, Wc2, bc2,
      cnt, mhat, msum, nE);

  egnn_node_kernel<<<(nN + 127) / 128, 256, 0, stream>>>(
      x, Wl1, bl1, Wl2, bl2, cnt, mhat, msum, out, nN);
}
